// ComplexityDecoderLayer_39968965656880
// MI455X (gfx1250) — compile-verified, matrix operand order checked by reading, not run on hardware
//
#include <hip/hip_runtime.h>
#include <math.h>

// ---------------- problem dims ----------------
#define Tn   2048
#define Dn   2048
#define Hn   16
#define KHn  4
#define HDn  128
#define In   5632
#define En   4
#define CHn  64
#define DTc  0.1f
#define EPSc 1e-6f
#define SCALEc 0.08838834764831845f   // HD^-0.5
#define LN10000 9.210340371976184f

typedef __bf16 bf16_t;
typedef bf16_t v8bf  __attribute__((ext_vector_type(8)));
typedef bf16_t v16bf __attribute__((ext_vector_type(16)));
typedef float  v8f   __attribute__((ext_vector_type(8)));

// ---------------- WMMA helpers ----------------
// 16x32 bf16 fragment (A layout; B 32x16 is lane-symmetric per ISA 7.12.2):
// lane L: row/col = L%16 ; element e -> K = (e<8?0:16) + 8*(L/16) + (e&7)
__device__ __forceinline__ v16bf load_frag(const bf16_t* rowp, int half) {
  v8bf lo = *(const v8bf*)(rowp + half * 8);
  v8bf hi = *(const v8bf*)(rowp + 16 + half * 8);
  v16bf r;
#pragma unroll
  for (int i = 0; i < 8; ++i) { r[i] = lo[i]; r[i + 8] = hi[i]; }
  return r;
}

__device__ __forceinline__ v8f wmma_bf16(v16bf a, v16bf b, v8f c) {
  return __builtin_amdgcn_wmma_f32_16x16x32_bf16(false, a, false, b, (short)0, c,
                                                 false, false);
}

// ---------------- generic WMMA GEMM ----------------
// C[M,N] (+)= A1[M,K](bf16) @ B1[K,N](f32->bf16 on the fly)
//           + (optional) A2 @ B2  (same shapes)
// Optional row-gated epilogue (MoE dispatch) and bf16 shadow output.
// Requires M%128==0, N%128==0, K%32==0.
// block = 512 threads (16 waves), macro-tile 128x128, each wave owns 32x32
// (2 A-frags x 2 B-frags -> 4 WMMAs per k-step; ~80 live VGPRs, no spills).
__global__ void gemm_bf16_wmma(const bf16_t* __restrict__ A1, const float* __restrict__ B1,
                               const bf16_t* __restrict__ A2, const float* __restrict__ B2,
                               float* __restrict__ C, bf16_t* __restrict__ Cb,
                               int M, int N, int K,
                               const float* __restrict__ rowgate,
                               const int* __restrict__ roweidx,
                               int expert, int accumulate) {
  (void)M;
  const int tid  = threadIdx.x;
  const int lane = tid & 31, wid = tid >> 5;
  const int wm   = wid >> 2, wn = wid & 3;          // 4x4 wave grid
  const int half = lane >> 4, lm = lane & 15;
  const int bM = blockIdx.y * 128, bN = blockIdx.x * 128;

  __shared__ __align__(16) bf16_t As[128][40];      // [m][k], padded
  __shared__ __align__(16) bf16_t Bs[128][40];      // [n][k] (transposed), padded

  v8f acc[2][2];
#pragma unroll
  for (int i = 0; i < 2; ++i)
#pragma unroll
    for (int j = 0; j < 2; ++j)
#pragma unroll
      for (int r = 0; r < 8; ++r) acc[i][j][r] = 0.f;

  // per-thread staging coordinates
  const int arow = tid >> 2;          // 0..127
  const int ac8  = (tid & 3) * 8;     // 0,8,16,24
  const int bkp  = tid >> 5;          // k-pair 0..15
  const int bnc  = (tid & 31) * 4;    // 4-col group 0..124

  const int phases = (A2 != nullptr) ? 2 : 1;
  for (int ph = 0; ph < phases; ++ph) {
    const bf16_t* A = ph ? A2 : A1;
    const float*  B = ph ? B2 : B1;
    for (int k0 = 0; k0 < K; k0 += 32) {
      __syncthreads();
      // A tile: 128x32 bf16, one 16B load per thread
      *(v8bf*)(&As[arow][ac8]) =
          *(const v8bf*)(A + (size_t)(bM + arow) * K + k0 + ac8);
      // B tile: 32x128 f32 -> bf16 transposed to [n][k];
      // two consecutive k rows packed -> 32-bit DS stores.
      {
        const float* b0 = B + (size_t)(k0 + bkp * 2) * N + bN + bnc;
        float4 f0 = *(const float4*)(b0);
        float4 f1 = *(const float4*)(b0 + N);
        if (k0 + 32 < K) __builtin_prefetch(b0 + (size_t)32 * N, 0, 3); // next K-tile
        union { bf16_t h[2]; unsigned u32; } p;
        p.h[0] = (bf16_t)f0.x; p.h[1] = (bf16_t)f1.x;
        *(unsigned*)(&Bs[bnc + 0][bkp * 2]) = p.u32;
        p.h[0] = (bf16_t)f0.y; p.h[1] = (bf16_t)f1.y;
        *(unsigned*)(&Bs[bnc + 1][bkp * 2]) = p.u32;
        p.h[0] = (bf16_t)f0.z; p.h[1] = (bf16_t)f1.z;
        *(unsigned*)(&Bs[bnc + 2][bkp * 2]) = p.u32;
        p.h[0] = (bf16_t)f0.w; p.h[1] = (bf16_t)f1.w;
        *(unsigned*)(&Bs[bnc + 3][bkp * 2]) = p.u32;
      }
      __syncthreads();
      v16bf a0 = load_frag(&As[wm * 32 + lm][0], half);
      v16bf a1 = load_frag(&As[wm * 32 + 16 + lm][0], half);
      v16bf b0 = load_frag(&Bs[wn * 32 + lm][0], half);
      v16bf b1 = load_frag(&Bs[wn * 32 + 16 + lm][0], half);
      acc[0][0] = wmma_bf16(a0, b0, acc[0][0]);
      acc[0][1] = wmma_bf16(a0, b1, acc[0][1]);
      acc[1][0] = wmma_bf16(a1, b0, acc[1][0]);
      acc[1][1] = wmma_bf16(a1, b1, acc[1][1]);
    }
  }

#pragma unroll
  for (int i = 0; i < 2; ++i)
#pragma unroll
    for (int j = 0; j < 2; ++j)
#pragma unroll
      for (int r = 0; r < 8; ++r) { // C layout: VGPR r -> m = 8*half + r, n = lm
        int m = bM + wm * 32 + i * 16 + half * 8 + r;
        int n = bN + wn * 32 + j * 16 + lm;
        float v = acc[i][j][r];
        if (rowgate) {
          float g = (roweidx[m] == expert) ? rowgate[m] : 0.f;
          v *= g;
        }
        size_t idx = (size_t)m * N + n;
        if (accumulate) v += C[idx];
        C[idx] = v;
        if (Cb) Cb[idx] = (bf16_t)v;
      }
}

// ---------------- flash attention (GQA, causal) ----------------
// grid (T/16, H), block 32 (one wave). 16 queries x all keys in 32-key steps.
__global__ void flash_attn_kernel(const bf16_t* __restrict__ qb,
                                  const bf16_t* __restrict__ kb,
                                  const bf16_t* __restrict__ vb,
                                  float* __restrict__ attnh,
                                  bf16_t* __restrict__ attnhb) {
  const int qt0 = blockIdx.x * 16;
  const int h   = blockIdx.y;
  const int kh  = h / (Hn / KHn);
  const int lane = threadIdx.x;
  const int half = lane >> 4, lm = lane & 15;

  __shared__ __align__(16) bf16_t Qs[16][HDn];
  __shared__ __align__(16) bf16_t Ks[32][HDn];
  __shared__ __align__(16) bf16_t Vt[HDn][32];   // transposed: [hd][key]
  __shared__ __align__(16) bf16_t Ps[16][32];

#pragma unroll
  for (int i = 0; i < 8; ++i) { // load Q tile 16x128
    int c = lane + 32 * i;
    int row = c >> 4, col8 = (c & 15) * 8;
    *(v8bf*)(&Qs[row][col8]) =
        *(const v8bf*)(qb + (size_t)(qt0 + row) * (Hn * HDn) + h * HDn + col8);
  }
  __syncthreads();

  v8f o[8];
#pragma unroll
  for (int nt = 0; nt < 8; ++nt)
#pragma unroll
    for (int r = 0; r < 8; ++r) o[nt][r] = 0.f;
  float mrow[8], lrow[8];
#pragma unroll
  for (int r = 0; r < 8; ++r) { mrow[r] = -1e30f; lrow[r] = 0.f; }

  const int jend = qt0 + 16;
  for (int j0 = 0; j0 < jend; j0 += 32) {
    __syncthreads();
#pragma unroll
    for (int i = 0; i < 16; ++i) { // load K (row-major) and V (transposed)
      int c = lane + 32 * i;
      int row = c >> 4, col8 = (c & 15) * 8;
      *(v8bf*)(&Ks[row][col8]) =
          *(const v8bf*)(kb + (size_t)(j0 + row) * (KHn * HDn) + kh * HDn + col8);
      v8bf vv = *(const v8bf*)(vb + (size_t)(j0 + row) * (KHn * HDn) + kh * HDn + col8);
#pragma unroll
      for (int q = 0; q < 8; ++q) Vt[col8 + q][row] = vv[q];
    }
    __syncthreads();

    v8f s0 = {0.f,0.f,0.f,0.f,0.f,0.f,0.f,0.f};
    v8f s1 = {0.f,0.f,0.f,0.f,0.f,0.f,0.f,0.f};
#pragma unroll
    for (int kc = 0; kc < 4; ++kc) {               // S = Q K^T over HD=128
      v16bf aq = load_frag(&Qs[lm][kc * 32], half);
      v16bf b0 = load_frag(&Ks[lm][kc * 32], half);
      v16bf b1 = load_frag(&Ks[lm + 16][kc * 32], half);
      s0 = wmma_bf16(aq, b0, s0);
      s1 = wmma_bf16(aq, b1, s1);
    }

    float corr[8];
#pragma unroll
    for (int r = 0; r < 8; ++r) {                  // online softmax per row
      int mg = qt0 + half * 8 + r;
      float a0 = (j0 + lm      <= mg) ? s0[r] * SCALEc : -1e30f;
      float a1 = (j0 + 16 + lm <= mg) ? s1[r] * SCALEc : -1e30f;
      float mx = fmaxf(a0, a1);
      mx = fmaxf(mx, __shfl_xor(mx, 1, 32));
      mx = fmaxf(mx, __shfl_xor(mx, 2, 32));
      mx = fmaxf(mx, __shfl_xor(mx, 4, 32));
      mx = fmaxf(mx, __shfl_xor(mx, 8, 32));
      float mnew = fmaxf(mrow[r], mx);
      float c  = __expf(mrow[r] - mnew);
      float p0 = __expf(a0 - mnew);
      float p1 = __expf(a1 - mnew);
      float rs = p0 + p1;
      rs += __shfl_xor(rs, 1, 32);
      rs += __shfl_xor(rs, 2, 32);
      rs += __shfl_xor(rs, 4, 32);
      rs += __shfl_xor(rs, 8, 32);
      lrow[r] = lrow[r] * c + rs;
      mrow[r] = mnew;
      corr[r] = c;
      Ps[half * 8 + r][lm]      = (bf16_t)p0;
      Ps[half * 8 + r][lm + 16] = (bf16_t)p1;
    }
    __syncthreads();

    v16bf ap = load_frag(&Ps[lm][0], half);        // O = corr*O + P V
#pragma unroll
    for (int nt = 0; nt < 8; ++nt) {
#pragma unroll
      for (int r = 0; r < 8; ++r) o[nt][r] *= corr[r];
      v16bf bv = load_frag(&Vt[nt * 16 + lm][0], half);
      o[nt] = wmma_bf16(ap, bv, o[nt]);
    }
  }

#pragma unroll
  for (int nt = 0; nt < 8; ++nt)
#pragma unroll
    for (int r = 0; r < 8; ++r) {
      int mg  = qt0 + half * 8 + r;
      int col = h * HDn + nt * 16 + lm;
      float val = o[nt][r] / lrow[r];
      attnh[(size_t)mg * (Hn * HDn) + col]  = val;
      attnhb[(size_t)mg * (Hn * HDn) + col] = (bf16_t)val;
    }
}

// ---------------- pointwise / small kernels ----------------
__global__ void rmsnorm_to_bf16(const float* __restrict__ in, const float* __restrict__ w,
                                bf16_t* __restrict__ out, int cols) {
  int row = blockIdx.x, tid = threadIdx.x;
  __shared__ float red[256];
  const float* rp = in + (size_t)row * cols;
  float s = 0.f;
  for (int c = tid; c < cols; c += 256) { float v = rp[c]; s += v * v; }
  red[tid] = s; __syncthreads();
  for (int st = 128; st > 0; st >>= 1) { if (tid < st) red[tid] += red[tid + st]; __syncthreads(); }
  float scale = rsqrtf(red[0] / cols + EPSc);
  for (int c = tid; c < cols; c += 256)
    out[(size_t)row * cols + c] = (bf16_t)(rp[c] * scale * w[c]);
}

__global__ void f32_to_bf16(const float* __restrict__ in, bf16_t* __restrict__ out, size_t n) {
  size_t i = (size_t)blockIdx.x * blockDim.x + threadIdx.x;
  if (i < n) out[i] = (bf16_t)in[i];
}

// per-(t, head) QK rmsnorm + RoPE, writes bf16. grid (T, H+KH), block HD
__global__ void qknorm_rope(const float* __restrict__ qf, const float* __restrict__ kf,
                            const int* __restrict__ positions,
                            const float* __restrict__ qn_w, const float* __restrict__ kn_w,
                            bf16_t* __restrict__ qb, bf16_t* __restrict__ kb) {
  int t = blockIdx.x, head = blockIdx.y, d = threadIdx.x;
  const float* src; const float* w; bf16_t* dst;
  if (head < Hn) {
    src = qf + (size_t)t * (Hn * HDn) + head * HDn; w = qn_w;
    dst = qb + (size_t)t * (Hn * HDn) + head * HDn;
  } else {
    int khh = head - Hn;
    src = kf + (size_t)t * (KHn * HDn) + khh * HDn; w = kn_w;
    dst = kb + (size_t)t * (KHn * HDn) + khh * HDn;
  }
  __shared__ float buf[HDn];
  __shared__ float red[HDn];
  float v = src[d];
  red[d] = v * v; __syncthreads();
  for (int st = 64; st > 0; st >>= 1) { if (d < st) red[d] += red[d + st]; __syncthreads(); }
  float scale = rsqrtf(red[0] / HDn + EPSc);
  buf[d] = v * scale * w[d];
  __syncthreads();
  if (d < HDn / 2) {
    float pos = (float)positions[t];
    float inv = __expf(-LN10000 * (2.f * d / HDn));
    float ang = pos * inv;
    float c = __cosf(ang), s = __sinf(ang);
    float x1 = buf[d], x2 = buf[d + HDn / 2];
    dst[d]           = (bf16_t)(x1 * c - x2 * s);
    dst[d + HDn / 2] = (bf16_t)(x2 * c + x1 * s);
  }
}

// c1 = tanh(ao @ Wc1 + bc1)   grid T, block CH
__global__ void c1_kernel(const float* __restrict__ ao, const float* __restrict__ Wc1,
                          const float* __restrict__ bc1, float* __restrict__ c1buf) {
  int t = blockIdx.x, j = threadIdx.x;
  float acc = 0.f;
  for (int k = 0; k < Dn; ++k) acc += ao[(size_t)t * Dn + k] * Wc1[(size_t)k * CHn + j];
  c1buf[(size_t)t * CHn + j] = tanhf(acc + bc1[j]);
}

// ctrl = sigmoid(c1 @ Wc2 + bc2); v_new & h1. grid T, block 256
__global__ void vnew_h1_kernel(const float* __restrict__ c1buf, const float* __restrict__ Wc2,
                               const float* __restrict__ bc2, const float* __restrict__ ao,
                               const float* __restrict__ mu, const float* __restrict__ velocity,
                               const float* __restrict__ hidden_in,
                               float* __restrict__ vnew_out, float* __restrict__ h1) {
  int t = blockIdx.x, tid = threadIdx.x;
  __shared__ float c1s[CHn];
  if (tid < CHn) c1s[tid] = c1buf[(size_t)t * CHn + tid];
  __syncthreads();
  for (int j = tid; j < Dn; j += 256) {
    float a = 0.f, b = 0.f;
    for (int k = 0; k < CHn; ++k) {
      float cv = c1s[k];
      a += cv * Wc2[(size_t)k * (2 * Dn) + j];
      b += cv * Wc2[(size_t)k * (2 * Dn) + Dn + j];
    }
    float alpha = 1.f / (1.f + __expf(-(a + bc2[j])));
    float beta  = 1.f / (1.f + __expf(-(b + bc2[Dn + j])));
    size_t idx = (size_t)t * Dn + j;
    float aov = ao[idx], muv = mu[idx], vel = velocity[idx];
    float vn = vel + DTc * (alpha * (muv - aov) - beta * vel);
    vnew_out[idx] = vn;
    h1[idx] = hidden_in[idx] + aov + DTc * vn;
  }
}

// router: logits = router_table[tok] + mu @ Wmu_router; argmax + softmax gate
__global__ void router_kernel(const float* __restrict__ mu, const float* __restrict__ Wmu_router,
                              const float* __restrict__ router_table,
                              const int* __restrict__ token_ids,
                              float* __restrict__ gate, int* __restrict__ eidx) {
  int t = blockIdx.x, tid = threadIdx.x;
  __shared__ float red[256];
  __shared__ float lsh[En];
  for (int e = 0; e < En; ++e) {
    float s = 0.f;
    for (int k = tid; k < Dn; k += 256)
      s += mu[(size_t)t * Dn + k] * Wmu_router[(size_t)k * En + e];
    red[tid] = s; __syncthreads();
    for (int st = 128; st > 0; st >>= 1) { if (tid < st) red[tid] += red[tid + st]; __syncthreads(); }
    if (tid == 0) lsh[e] = red[0];
    __syncthreads();
  }
  if (tid == 0) {
    int tok = token_ids[t];
    float lv[En]; float mx = -1e30f; int best = 0;
    for (int e = 0; e < En; ++e) {
      lv[e] = lsh[e] + router_table[(size_t)tok * En + e];
      if (lv[e] > mx) { mx = lv[e]; best = e; }
    }
    float sum = 0.f;
    for (int e = 0; e < En; ++e) sum += __expf(lv[e] - mx);
    gate[t] = __expf(lv[best] - mx) / sum;
    eidx[t] = best;
  }
}

__global__ void silu_mul_kernel(const float* __restrict__ g, const float* __restrict__ u,
                                bf16_t* __restrict__ out, size_t n) {
  size_t i = (size_t)blockIdx.x * blockDim.x + threadIdx.x;
  if (i < n) {
    float gv = g[i];
    out[i] = (bf16_t)((gv / (1.f + __expf(-gv))) * u[i]);
  }
}

__global__ void add_kernel(const float* __restrict__ a, const float* __restrict__ b,
                           float* __restrict__ out, size_t n) {
  size_t i = (size_t)blockIdx.x * blockDim.x + threadIdx.x;
  if (i < n) out[i] = a[i] + b[i];
}

// ---------------- launch ----------------
extern "C" void kernel_launch(void* const* d_in, const int* in_sizes, int n_in,
                              void* d_out, int out_size, void* d_ws, size_t ws_size,
                              hipStream_t stream) {
  (void)in_sizes; (void)n_in; (void)out_size; (void)ws_size;
  const int*   positions = (const int*)  d_in[0];
  const float* hidden    = (const float*)d_in[1];
  const float* velocity  = (const float*)d_in[2];
  const int*   token_ids = (const int*)  d_in[3];
  const float* mu_prev   = (const float*)d_in[4];
  const float* ln1_w     = (const float*)d_in[5];
  const float* ln2_w     = (const float*)d_in[6];
  const float* Wq        = (const float*)d_in[7];
  const float* Wk        = (const float*)d_in[8];
  const float* Wv        = (const float*)d_in[9];
  const float* Wo        = (const float*)d_in[10];
  const float* Wmu_q     = (const float*)d_in[11];
  const float* Wmu_k     = (const float*)d_in[12];
  const float* Wmu_v     = (const float*)d_in[13];
  const float* qn_w      = (const float*)d_in[14];
  const float* kn_w      = (const float*)d_in[15];
  const float* Wdyn_mu   = (const float*)d_in[16];
  const float* Wc1       = (const float*)d_in[17];
  const float* bc1       = (const float*)d_in[18];
  const float* Wc2       = (const float*)d_in[19];
  const float* bc2       = (const float*)d_in[20];
  const float* router_table = (const float*)d_in[21];
  const float* Wmu_router   = (const float*)d_in[22];
  const float* Wg        = (const float*)d_in[23];
  const float* Wu        = (const float*)d_in[24];
  const float* Wd        = (const float*)d_in[25];

  float* out_hidden = (float*)d_out;
  float* out_vnew   = out_hidden + (size_t)Tn * Dn;
  float* out_mu     = out_vnew   + (size_t)Tn * Dn;

  char* wsp = (char*)d_ws;
  size_t off = 0;
  auto alloc = [&](size_t bytes) -> void* {
    void* p = wsp + off;
    off += (bytes + 255) & ~(size_t)255;
    return p;
  };
  bf16_t* xb    = (bf16_t*)alloc((size_t)Tn * Dn * 2);
  bf16_t* mub   = (bf16_t*)alloc((size_t)Tn * Dn * 2);
  float*  qf    = (float*) alloc((size_t)Tn * Hn * HDn * 4);
  float*  kf    = (float*) alloc((size_t)Tn * KHn * HDn * 4);
  float*  vf    = (float*) alloc((size_t)Tn * KHn * HDn * 4);
  bf16_t* qbb   = (bf16_t*)alloc((size_t)Tn * Hn * HDn * 2);
  bf16_t* kbb   = (bf16_t*)alloc((size_t)Tn * KHn * HDn * 2);
  bf16_t* vbb   = (bf16_t*)alloc((size_t)Tn * KHn * HDn * 2);
  bf16_t* attnhb= (bf16_t*)alloc((size_t)Tn * Hn * HDn * 2);
  float*  ao    = (float*) alloc((size_t)Tn * Dn * 4);
  bf16_t* aob   = (bf16_t*)alloc((size_t)Tn * Dn * 2);
  float*  c1buf = (float*) alloc((size_t)Tn * CHn * 4);
  float*  h1    = (float*) alloc((size_t)Tn * Dn * 4);
  bf16_t* x2b   = (bf16_t*)alloc((size_t)Tn * Dn * 2);
  float*  gate  = (float*) alloc((size_t)Tn * 4);
  int*    eidx  = (int*)   alloc((size_t)Tn * 4);
  float*  gbuf  = (float*) alloc((size_t)Tn * In * 4);
  float*  ubuf  = (float*) alloc((size_t)Tn * In * 4);
  bf16_t* actb  = (bf16_t*)alloc((size_t)Tn * In * 2);
  float*  outacc= (float*) alloc((size_t)Tn * Dn * 4);
  float*  attnh = qf;  // qf is dead after rope; reuse for attention output

  const size_t TD = (size_t)Tn * Dn;
  const size_t TKV = (size_t)Tn * KHn * HDn;
  const size_t TI = (size_t)Tn * In;

  // 1) x = rmsnorm(hidden, ln1) -> bf16 ; mu_prev -> bf16
  rmsnorm_to_bf16<<<Tn, 256, 0, stream>>>(hidden, ln1_w, xb, Dn);
  f32_to_bf16<<<(int)((TD + 255) / 256), 256, 0, stream>>>(mu_prev, mub, TD);

  // 2) dual-input QKV projections (WMMA)
  gemm_bf16_wmma<<<dim3((Hn * HDn) / 128, Tn / 128), 512, 0, stream>>>(
      xb, Wq, mub, Wmu_q, qf, nullptr, Tn, Hn * HDn, Dn, nullptr, nullptr, 0, 0);
  gemm_bf16_wmma<<<dim3((KHn * HDn) / 128, Tn / 128), 512, 0, stream>>>(
      xb, Wk, mub, Wmu_k, kf, nullptr, Tn, KHn * HDn, Dn, nullptr, nullptr, 0, 0);
  gemm_bf16_wmma<<<dim3((KHn * HDn) / 128, Tn / 128), 512, 0, stream>>>(
      xb, Wv, mub, Wmu_v, vf, nullptr, Tn, KHn * HDn, Dn, nullptr, nullptr, 0, 0);

  // 3) per-head QK rmsnorm + RoPE -> bf16 ; V -> bf16
  qknorm_rope<<<dim3(Tn, Hn + KHn), HDn, 0, stream>>>(qf, kf, positions, qn_w, kn_w, qbb, kbb);
  f32_to_bf16<<<(int)((TKV + 255) / 256), 256, 0, stream>>>(vf, vbb, TKV);

  // 4) causal GQA flash attention (WMMA)
  flash_attn_kernel<<<dim3(Tn / 16, Hn), 32, 0, stream>>>(qbb, kbb, vbb, attnh, attnhb);

  // 5) ao = attn @ Wo ; mu = ao @ Wdyn_mu (-> output slot 3)
  gemm_bf16_wmma<<<dim3(Dn / 128, Tn / 128), 512, 0, stream>>>(
      attnhb, Wo, nullptr, nullptr, ao, aob, Tn, Dn, Hn * HDn, nullptr, nullptr, 0, 0);
  gemm_bf16_wmma<<<dim3(Dn / 128, Tn / 128), 512, 0, stream>>>(
      aob, Wdyn_mu, nullptr, nullptr, out_mu, nullptr, Tn, Dn, Dn, nullptr, nullptr, 0, 0);

  // 6) control path -> v_new (output slot 2) and h1
  c1_kernel<<<Tn, CHn, 0, stream>>>(ao, Wc1, bc1, c1buf);
  vnew_h1_kernel<<<Tn, 256, 0, stream>>>(c1buf, Wc2, bc2, ao, out_mu, velocity,
                                         hidden, out_vnew, h1);

  // 7) second rmsnorm, router
  rmsnorm_to_bf16<<<Tn, 256, 0, stream>>>(h1, ln2_w, x2b, Dn);
  router_kernel<<<Tn, 256, 0, stream>>>(out_mu, Wmu_router, router_table, token_ids, gate, eidx);

  // 8) MoE: dense per expert (matches reference compute-then-mask), gated epilogue
  for (int e = 0; e < En; ++e) {
    const float* Wge = Wg + (size_t)e * Dn * In;
    const float* Wue = Wu + (size_t)e * Dn * In;
    const float* Wde = Wd + (size_t)e * In * Dn;
    gemm_bf16_wmma<<<dim3(In / 128, Tn / 128), 512, 0, stream>>>(
        x2b, Wge, nullptr, nullptr, gbuf, nullptr, Tn, In, Dn, nullptr, nullptr, 0, 0);
    gemm_bf16_wmma<<<dim3(In / 128, Tn / 128), 512, 0, stream>>>(
        x2b, Wue, nullptr, nullptr, ubuf, nullptr, Tn, In, Dn, nullptr, nullptr, 0, 0);
    silu_mul_kernel<<<(int)((TI + 255) / 256), 256, 0, stream>>>(gbuf, ubuf, actb, TI);
    gemm_bf16_wmma<<<dim3(Dn / 128, Tn / 128), 512, 0, stream>>>(
        actb, Wde, nullptr, nullptr, outacc, nullptr, Tn, Dn, In,
        gate, eidx, e, (e > 0) ? 1 : 0);
  }

  // 9) final residual add -> output slot 1
  add_kernel<<<(int)((TD + 255) / 256), 256, 0, stream>>>(h1, outacc, out_hidden, TD);
}
